// DeepKernelNN_2740189135781
// MI455X (gfx1250) — matrix operand order repaired and checked by reading, not compiled, not run on hardware
//
#include <hip/hip_runtime.h>

// ---------------- problem constants (from reference) ----------------
#define NN     4096      // nodes
#define EE     32768     // edges
#define WIDTH  64
#define KERW   1024
#define KHALF  512       // KER_W/2
#define DEPTH  4
#define KIN    6

// ---------------- CDNA5 WMMA types ----------------
typedef __attribute__((ext_vector_type(16))) __bf16 v16bf;
typedef __attribute__((ext_vector_type(8)))  __bf16 v8bf;
typedef __attribute__((ext_vector_type(8)))  float  v8f;

__device__ __forceinline__ unsigned short f2bfu(float f) {
  unsigned u = __float_as_uint(f);
  u += 0x7FFFu + ((u >> 16) & 1u);            // round-to-nearest-even
  return (unsigned short)(u >> 16);
}

__device__ __forceinline__ v8f wmma_bf16(v16bf a, v16bf b, v8f c) {
  // (neg_a, A, neg_b, B, c_mod, C, reuse_a, reuse_b)
  return __builtin_amdgcn_wmma_f32_16x16x32_bf16(false, a, false, b, (short)0, c,
                                                 false, false);
}

// A fragment (16x32 bf16) from a row-major bf16 tile, row stride ld elements.
// Order used here (not the ISA table order): lanes 0-15 row m=lane, K = kh*16+0..15 (contiguous), kh = lane>>4.
__device__ __forceinline__ v16bf load_a_frag(const unsigned short* tile, int ld,
                                             int rowBase, int k0, int lane) {
  int m  = rowBase + (lane & 15);
  int kh = (lane >> 4) * 16;
  v8bf lo = *(const v8bf*)(tile + (size_t)m * ld + k0 + kh);
  v8bf hi = *(const v8bf*)(tile + (size_t)m * ld + k0 + 8 + kh);
  v16bf a;
#pragma unroll
  for (int j = 0; j < 8; ++j) { a[j] = lo[j]; a[8 + j] = hi[j]; }
  return a;
}

// B fragment (32x16 bf16) from a PRE-TRANSPOSED weight wT[n][K] (ld = K length).
// Order used here (matches load_a_frag): lane n%16 = column, K = (lane>>4)*16 + j : 16 contiguous bf16.
__device__ __forceinline__ v16bf load_b_frag(const unsigned short* wT, size_t rowBase,
                                             int ld, int k0, int lane) {
  size_t n = rowBase + (lane & 15);
  int kh = (lane >> 4) * 16;
  return *(const v16bf*)(wT + n * (size_t)ld + k0 + kh);
}

// ---------------- one-time (per call) weight conversion / transpose ----------------
__global__ void conv_kw2t(const float* __restrict__ kw2, unsigned short* __restrict__ kw2t) {
  int tid = blockIdx.x * blockDim.x + threadIdx.x;   // D*1024*512 = 2^21
  int d   = tid >> 19;
  int rem = tid & ((1 << 19) - 1);
  int n   = rem >> 9;
  int kk  = rem & 511;
  kw2t[tid] = f2bfu(kw2[(((size_t)d * KHALF) + kk) * KERW + n]);
}

__global__ void conv_kw3t(const float* __restrict__ kw3, unsigned short* __restrict__ kw3t) {
  size_t tid = (size_t)blockIdx.x * blockDim.x + threadIdx.x;  // D*4096*1024 = 2^24
  int d      = (int)(tid >> 22);
  size_t rem = tid & ((1u << 22) - 1);
  int o      = (int)(rem >> 10);
  int kk     = (int)(rem & 1023);
  kw3t[tid] = f2bfu(kw3[(((size_t)d * KERW) + kk) * (WIDTH * WIDTH) + o]);
}

// ---------------- small supporting kernels ----------------
__global__ void zero_f32(float* __restrict__ p, int n) {
  int i = blockIdx.x * blockDim.x + threadIdx.x;
  if (i < n) p[i] = 0.f;
}

__global__ void deg_accum(const int* __restrict__ ei, float* __restrict__ deg) {
  int e = blockIdx.x * blockDim.x + threadIdx.x;
  if (e < EE) atomicAdd(deg + ei[EE + e], 1.0f);
}

__global__ void inv_deg_k(float* __restrict__ d, int n) {
  int v = blockIdx.x * blockDim.x + threadIdx.x;
  if (v < n) { float x = d[v]; d[v] = x > 0.f ? 1.0f / x : 0.f; }
}

__global__ void init_h(const float* __restrict__ x, const float* __restrict__ w,
                       const float* __restrict__ b, float* __restrict__ h) {
  int v = blockIdx.x, o = threadIdx.x;
  float acc = b[o];
#pragma unroll
  for (int j = 0; j < KIN; ++j) acc += x[(size_t)v * KIN + j] * w[j * WIDTH + o];
  h[(size_t)v * WIDTH + o] = acc;
}

__global__ void node_update(int k, const float* __restrict__ h, const float* __restrict__ agg,
                            const float* __restrict__ root, const float* __restrict__ bias,
                            float* __restrict__ hn) {
  __shared__ float hrow[WIDTH];
  int v = blockIdx.x, o = threadIdx.x;
  hrow[o] = h[(size_t)v * WIDTH + o];
  __syncthreads();
  float acc = bias[(size_t)k * WIDTH + o];
  const float* rt = root + (size_t)k * WIDTH * WIDTH;
#pragma unroll 8
  for (int i = 0; i < WIDTH; ++i) acc += hrow[i] * rt[i * WIDTH + o];
  float val = agg[(size_t)v * WIDTH + o] + acc;
  hn[(size_t)v * WIDTH + o] = fmaxf(val, 0.f);
}

__global__ void final_out(const float* __restrict__ h, const float* __restrict__ w,
                          const float* __restrict__ b, float* __restrict__ out) {
  int v = blockIdx.x * blockDim.x + threadIdx.x;
  if (v < NN) {
    float acc = b[0];
#pragma unroll 8
    for (int i = 0; i < WIDTH; ++i) acc += h[(size_t)v * WIDTH + i] * w[i];
    out[v] = acc;
  }
}

// ---------------- fused per-depth kernel ----------------
// One workgroup = 64 edges, 512 threads = 16 wave32s (4 M-tiles x 4 N-tiles).
// LDS: e1 tile 64x512 bf16 (64KB) + e2 tile 64x1024 bf16 (128KB)
//      + hsT 64x64 f32 (16KB) + edge_attr/meta  => 210 KB of 320 KB per WGP.
#define FUSED_LDS (64*KHALF*2 + 64*KERW*2 + 64*64*4 + 64*6*4 + 64*4 + 64*4)

__global__ __launch_bounds__(512, 1)
void fused_msg(int k, const int* __restrict__ ei, const float* __restrict__ eattr,
               const float* __restrict__ kw1, const float* __restrict__ kb1,
               const float* __restrict__ kb2, const float* __restrict__ kb3,
               const unsigned short* __restrict__ kw2t,
               const unsigned short* __restrict__ kw3t,
               const float* __restrict__ invdeg, const float* __restrict__ h,
               float* __restrict__ agg) {
  extern __shared__ char smem[];
  unsigned short* e1 = (unsigned short*)smem;        // [64][512] bf16
  unsigned short* e2 = e1 + 64 * KHALF;              // [64][1024] bf16
  float* hsT  = (float*)(e2 + 64 * KERW);            // [64 ch][64 edges] f32
  float* ea   = hsT + 64 * 64;                       // [64][6]
  float* idg  = ea + 64 * 6;                         // [64]
  int*   dstl = (int*)(idg + 64);                    // [64]

  const int t    = threadIdx.x;
  const int lane = t & 31;
  const int wave = t >> 5;
  const int eb   = blockIdx.x * 64;

  // ---- Phase 0: tile metadata + gather h[src] ----
  if (t < 64) {
    int d = ei[EE + eb + t];
    dstl[t] = d;
    idg[t]  = invdeg[d];
  }
  if (t < 64 * KIN) ea[t] = eattr[(size_t)eb * KIN + t];
#pragma unroll
  for (int r = 0; r < 8; ++r) {
    int idx = t + r * 512;              // 0..4095
    int el = idx >> 6, c = idx & 63;
    int s = ei[eb + el];
    hsT[c * 64 + el] = h[(size_t)s * WIDTH + c];   // transposed for phase 3
  }
  __syncthreads();

  // ---- Phase 1: e1 = relu(edge_attr @ kw1 + kb1) -> LDS bf16 ----
  {
    const float* w1 = kw1 + (size_t)k * KIN * KHALF;
    float w0 = w1[0 * KHALF + t], wv1 = w1[1 * KHALF + t], w2 = w1[2 * KHALF + t];
    float w3 = w1[3 * KHALF + t], w4 = w1[4 * KHALF + t], w5 = w1[5 * KHALF + t];
    float b1 = kb1[(size_t)k * KHALF + t];
#pragma unroll 4
    for (int e = 0; e < 64; ++e) {
      const float* a = ea + e * KIN;
      float acc = b1 + a[0] * w0 + a[1] * wv1 + a[2] * w2 + a[3] * w3 + a[4] * w4 + a[5] * w5;
      e1[e * KHALF + t] = f2bfu(fmaxf(acc, 0.f));
    }
  }
  __syncthreads();

  // ---- Phase 2: e2 = relu(e1 @ kw2 + kb2) via WMMA -> LDS bf16 ----
#pragma unroll 1
  for (int it = 0; it < 16; ++it) {
    int tile = it * 16 + wave;              // 256 tiles: 4 mt x 64 nt
    int mt = (tile & 3) * 16;
    int nt = (tile >> 2) * 16;
    float bb = kb2[(size_t)k * KERW + nt + (lane & 15)];
    v8f c;
#pragma unroll
    for (int r = 0; r < 8; ++r) c[r] = bb;
#pragma unroll 4
    for (int ks = 0; ks < KHALF / 32; ++ks) {
      v16bf a = load_a_frag(e1, KHALF, mt, ks * 32, lane);
      v16bf b = load_b_frag(kw2t, (size_t)k * KERW + nt, KHALF, ks * 32, lane);
      c = wmma_bf16(a, b, c);
    }
    int mrow = mt + ((lane >> 4) << 3);
    int ncol = nt + (lane & 15);
#pragma unroll
    for (int r = 0; r < 8; ++r)
      e2[(mrow + r) * KERW + ncol] = f2bfu(fmaxf(c[r], 0.f));
  }
  __syncthreads();

  // ---- Phase 3: stream input channels in PAIRS: one shared A feeds two WMMAs ----
  // msg[e,o] = sum_i h_src[e,i] * ( e2[e,:] @ kw3[:, i*64+o] + kb3[i*64+o] )
  {
    int mt   = (wave & 3) * 16;
    int nt   = (wave >> 2) * 16;           // 0..48
    int col  = nt + (lane & 15);           // output channel o in [0,64)
    int mrow = mt + ((lane >> 4) << 3);
    v8f msg;
#pragma unroll
    for (int r = 0; r < 8; ++r) msg[r] = 0.f;
    const unsigned short* w3base = kw3t + ((size_t)k * (WIDTH * WIDTH) + col) * KERW;
    const float* kb3base = kb3 + (size_t)k * (WIDTH * WIDTH);
    const int kh = (lane >> 4) * 16;
#pragma unroll 1
    for (int i = 0; i < WIDTH; i += 2) {
      float bb0 = kb3base[(i + 0) * WIDTH + col];   // exact fp32 bias path
      float bb1 = kb3base[(i + 1) * WIDTH + col];
      v8f c0, c1;
#pragma unroll
      for (int r = 0; r < 8; ++r) { c0[r] = bb0; c1[r] = bb1; }
      const unsigned short* wrow0 = w3base + (size_t)(i + 0) * WIDTH * KERW;
      const unsigned short* wrow1 = w3base + (size_t)(i + 1) * WIDTH * KERW;
      if (i < WIDTH - 2) {
        __builtin_prefetch(wrow0 + (size_t)2 * WIDTH * KERW, 0, 0);
        __builtin_prefetch(wrow1 + (size_t)2 * WIDTH * KERW, 0, 0);
      }
#pragma unroll 2
      for (int ks = 0; ks < KERW / 32; ++ks) {
        v16bf a  = load_a_frag(e2, KERW, mt, ks * 32, lane);   // loaded ONCE per pair
        v16bf b0 = *(const v16bf*)(wrow0 + ks * 32 + kh);
        v16bf b1 = *(const v16bf*)(wrow1 + ks * 32 + kh);
        c0 = wmma_bf16(a, b0, c0);   // two independent accumulation chains
        c1 = wmma_bf16(a, b1, c1);
      }
#pragma unroll
      for (int r = 0; r < 8; ++r) {
        msg[r] += hsT[(i + 0) * 64 + mrow + r] * c0[r]
                + hsT[(i + 1) * 64 + mrow + r] * c1[r];  // fp32 per-edge contraction
      }
    }
    // scatter-mean into agg
#pragma unroll
    for (int r = 0; r < 8; ++r) {
      int el = mrow + r;
      atomicAdd(agg + (size_t)dstl[el] * WIDTH + col, msg[r] * idg[el]);
    }
  }
}

// ---------------- host-side orchestration ----------------
extern "C" void kernel_launch(void* const* d_in, const int* in_sizes, int n_in,
                              void* d_out, int out_size, void* d_ws, size_t ws_size,
                              hipStream_t stream) {
  (void)in_sizes; (void)n_in; (void)out_size; (void)ws_size;
  const float* x     = (const float*)d_in[0];
  const float* eattr = (const float*)d_in[1];
  const float* fc1_w = (const float*)d_in[2];
  const float* fc1_b = (const float*)d_in[3];
  const float* kw1   = (const float*)d_in[4];
  const float* kb1   = (const float*)d_in[5];
  const float* kw2   = (const float*)d_in[6];
  const float* kb2   = (const float*)d_in[7];
  const float* kw3   = (const float*)d_in[8];
  const float* kb3   = (const float*)d_in[9];
  const float* root  = (const float*)d_in[10];
  const float* bias  = (const float*)d_in[11];
  const float* fc2_w = (const float*)d_in[12];
  const float* fc2_b = (const float*)d_in[13];
  const int*   ei    = (const int*)d_in[14];
  float* out = (float*)d_out;

  char* base = (char*)d_ws;
  size_t off = 0;
  auto carve = [&](size_t bytes) -> char* {
    char* p = base + off;
    off += (bytes + 255) & ~(size_t)255;
    return p;
  };
  float* hA   = (float*)carve((size_t)NN * WIDTH * 4);
  float* hB   = (float*)carve((size_t)NN * WIDTH * 4);
  float* agg  = (float*)carve((size_t)NN * WIDTH * 4);
  float* ideg = (float*)carve((size_t)NN * 4);
  unsigned short* kw2t = (unsigned short*)carve((size_t)DEPTH * KERW * KHALF * 2);
  unsigned short* kw3t = (unsigned short*)carve((size_t)DEPTH * (WIDTH * WIDTH) * KERW * 2);

  // weight conversion + transpose to bf16 (kw3t: 8 MB/depth -> L2 resident)
  conv_kw2t<<<(DEPTH * KERW * KHALF) / 256, 256, 0, stream>>>(kw2, kw2t);
  conv_kw3t<<<(DEPTH * (WIDTH * WIDTH) * KERW) / 256, 256, 0, stream>>>(kw3, kw3t);

  // degree -> inv_deg
  zero_f32<<<(NN + 255) / 256, 256, 0, stream>>>(ideg, NN);
  deg_accum<<<EE / 256, 256, 0, stream>>>(ei, ideg);
  inv_deg_k<<<(NN + 255) / 256, 256, 0, stream>>>(ideg, NN);

  // h = x @ fc1_w + fc1_b
  init_h<<<NN, WIDTH, 0, stream>>>(x, fc1_w, fc1_b, hA);

  float* hc = hA;
  float* hn = hB;
  for (int k = 0; k < DEPTH; ++k) {
    zero_f32<<<(NN * WIDTH) / 256, 256, 0, stream>>>(agg, NN * WIDTH);
    fused_msg<<<EE / 64, 512, FUSED_LDS, stream>>>(k, ei, eattr, kw1, kb1, kb2, kb3,
                                                   kw2t, kw3t, ideg, hc, agg);
    node_update<<<NN, WIDTH, 0, stream>>>(k, hc, agg, root, bias, hn);
    float* tmp = hc; hc = hn; hn = tmp;
  }

  final_out<<<(NN + 255) / 256, 256, 0, stream>>>(hc, fc2_w, fc2_b, out);
}
